// DreamEngine_4681514352757
// MI455X (gfx1250) — matrix operand order repaired and checked by reading, not compiled, not run on hardware
//
#include <hip/hip_runtime.h>
#include <cmath>

// ---------- types ----------
typedef __attribute__((ext_vector_type(16))) __bf16 v16bf;
typedef __attribute__((ext_vector_type(8)))  float  v8f;
typedef __attribute__((ext_vector_type(4)))  __bf16 v4bf;
typedef __attribute__((ext_vector_type(4)))  int    v4i;

#define AS1 __attribute__((address_space(1)))
#define AS3 __attribute__((address_space(3)))

#if defined(__has_builtin)
#if __has_builtin(__builtin_amdgcn_global_load_async_to_lds_b128)
#define HAVE_ASYNC_LDS 1
#endif
#endif
#ifndef HAVE_ASYNC_LDS
#define HAVE_ASYNC_LDS 0
#endif

// ---------- problem constants ----------
#define HD    4096
#define HALF_ 2048
#define COMP_ 1024
#define NS    16384           // NUM_SAMPLES
#define NFLAT 8192            // rows of hidden_states written into memory

// ---------- workspace layout (bytes) ----------
#define ALIGN256(x) (((x) + 255) & ~((size_t)255))
static constexpr size_t OFF_SAMP = 0;                                                   // [NS][HD] bf16
static constexpr size_t OFF_W1T  = ALIGN256(OFF_SAMP + (size_t)NS * HD * 2);            // [2048][4096] bf16
static constexpr size_t OFF_W2T  = ALIGN256(OFF_W1T  + (size_t)HALF_ * HD * 2);         // [2048][2048] bf16
static constexpr size_t OFF_WD1T = ALIGN256(OFF_W2T  + (size_t)HALF_ * HALF_ * 2);      // [2048][1024] bf16
static constexpr size_t OFF_WD2T = ALIGN256(OFF_WD1T + (size_t)HALF_ * COMP_ * 2);      // [4096][2048] bf16
static constexpr size_t OFF_H    = ALIGN256(OFF_WD2T + (size_t)HD * HALF_ * 2);         // [NS][2048] bf16
static constexpr size_t OFF_ENC  = ALIGN256(OFF_H    + (size_t)NS * HALF_ * 2);         // [NS][2048] f32
static constexpr size_t OFF_Z    = ALIGN256(OFF_ENC  + (size_t)NS * HALF_ * 4);         // [NS][1024] bf16
static constexpr size_t OFF_HD_  = ALIGN256(OFF_Z    + (size_t)NS * COMP_ * 2);         // [NS][2048] bf16
static constexpr size_t OFF_ACC  = ALIGN256(OFF_HD_  + (size_t)NS * HALF_ * 2);         // 2 f32 accumulators

// =====================================================================
// async copy helpers (CDNA5 GLOBAL_LOAD_ASYNC_TO_LDS, ASYNCcnt tracked)
// =====================================================================
__device__ __forceinline__ void async_copy_b128(const void* g, void* l) {
#if HAVE_ASYNC_LDS
    __builtin_amdgcn_global_load_async_to_lds_b128(
        (AS1 v4i*)(uintptr_t)g,
        (AS3 v4i*)(uint32_t)(uintptr_t)l, 0, 0);
#else
    *(uint4*)l = *(const uint4*)g;
#endif
}

__device__ __forceinline__ void wait_async_all() {
#if HAVE_ASYNC_LDS
#if defined(__has_builtin) && __has_builtin(__builtin_amdgcn_s_wait_asynccnt)
    __builtin_amdgcn_s_wait_asynccnt(0);
#else
    asm volatile("s_wait_asynccnt 0x0" ::: "memory");
#endif
#endif
}

// =====================================================================
// zero the global accumulators (recon sum, kl sum)
// =====================================================================
__global__ void zero_acc_kernel(float* acc) {
    if (threadIdx.x < 2) acc[threadIdx.x] = 0.0f;
}

// =====================================================================
// W[K][N] f32  ->  Wt[N][K] bf16   (tiled transpose, coalesced both ways)
// =====================================================================
__global__ __launch_bounds__(256) void transpose_bf16_kernel(
    const float* __restrict__ W, __bf16* __restrict__ Wt, int K, int N) {
    __shared__ float tile[32][33];
    const int tx = threadIdx.x & 31;
    const int ty = threadIdx.x >> 5;          // 0..7
    const int n0 = blockIdx.x * 32;
    const int k0 = blockIdx.y * 32;
#pragma unroll
    for (int i = 0; i < 32; i += 8)
        tile[ty + i][tx] = W[(size_t)(k0 + ty + i) * N + (n0 + tx)];
    __syncthreads();
#pragma unroll
    for (int i = 0; i < 32; i += 8)
        Wt[(size_t)(n0 + ty + i) * K + (k0 + tx)] = (__bf16)tile[tx][ty + i];
}

// =====================================================================
// samples_bf16[s][:] = bf16( idx[s] < 8192 ? hidden_flat[idx[s]] : memory[idx[s]] )
// =====================================================================
__global__ __launch_bounds__(256) void gather_samples_kernel(
    const float* __restrict__ hid, const float* __restrict__ mem,
    const int* __restrict__ idx, __bf16* __restrict__ out) {
    const int s = blockIdx.y;
    const int c = blockIdx.x * 1024 + threadIdx.x * 4;
    const int row = idx[s];
    const float* src = (row < NFLAT) ? (hid + (size_t)row * HD) : (mem + (size_t)row * HD);
    const float4 v = *(const float4*)(src + c);
    v4bf o = { (__bf16)v.x, (__bf16)v.y, (__bf16)v.z, (__bf16)v.w };
    *(v4bf*)(out + (size_t)s * HD + c) = o;
}

// =====================================================================
// WMMA GEMM:  C[M][N] = epilogue( A[M][K](bf16) @ Bt[N][K]^T(bf16) + bias )
// BM=BN=128, BK=64, 256 threads = 8 waves, each wave: 4x2 tiles of 16x16.
// Double-buffered LDS; tiles staged with GLOBAL_LOAD_ASYNC_TO_LDS_B128.
// MODE 0: out_bf = bf16(gelu(x))
// MODE 1: out_f  = x
// MODE 2: atomicAdd(accum, sum((x - samples_ref)^2))   (fused recon loss)
// =====================================================================
__device__ __forceinline__ float gelu_exact(float x) {
    return 0.5f * x * (1.0f + erff(x * 0.70710678118654752f));
}

#define LDA_ 72   // 64 bf16 payload + 8 bf16 pad (keeps 16B alignment, skews banks)

template <int MODE>
__global__ __launch_bounds__(256) void gemm_wmma_kernel(
    const __bf16* __restrict__ A, const __bf16* __restrict__ Bt,
    const float* __restrict__ bias, int M, int N, int K,
    __bf16* __restrict__ outBf, float* __restrict__ outF,
    const int* __restrict__ idx, const float* __restrict__ hid,
    const float* __restrict__ mem, float* __restrict__ accum) {

    __shared__ __align__(16) __bf16 sA[2][128 * LDA_];
    __shared__ __align__(16) __bf16 sB[2][128 * LDA_];

    const int tid  = threadIdx.x;
    const int lane = tid & 31;
    const int w    = tid >> 5;          // wave 0..7
    const int half = lane >> 4;         // 0/1
    const int l16  = lane & 15;
    const int wm   = (w & 1) * 64;      // wave row base in block tile
    const int wn   = (w >> 1) * 32;     // wave col base in block tile
    const int bm0  = blockIdx.y * 128;
    const int bn0  = blockIdx.x * 128;

    // per-thread staging coordinates: 4 x b128 per tile per matrix
    const int srow = tid >> 1;               // 0..127
    const int sch0 = (tid & 1) * 4;          // uint4 chunk base 0 or 4

    v8f acc[4][2];
#pragma unroll
    for (int i = 0; i < 4; ++i)
#pragma unroll
        for (int j = 0; j < 2; ++j) acc[i][j] = (v8f)0.0f;

    union Frag { v16bf v; uint4 q[2]; };

    const int niter = K >> 6;

    // ---- prologue: stage tile 0 into buffer 0 ----
#pragma unroll
    for (int i = 0; i < 4; ++i) {
        async_copy_b128(A  + (size_t)(bm0 + srow) * K + (sch0 + i) * 8,
                        &sA[0][srow * LDA_ + (sch0 + i) * 8]);
        async_copy_b128(Bt + (size_t)(bn0 + srow) * K + (sch0 + i) * 8,
                        &sB[0][srow * LDA_ + (sch0 + i) * 8]);
    }

    for (int it = 0; it < niter; ++it) {
        const int cur = it & 1;
        wait_async_all();        // my async stores for tile `it` have landed
        __syncthreads();         // everyone's landed; prior reads of `nxt` done

        // ---- issue async staging for tile it+1 while computing tile it ----
        if (it + 1 < niter) {
            const int nxt = cur ^ 1;
            const int kk  = (it + 1) << 6;
#pragma unroll
            for (int i = 0; i < 4; ++i) {
                async_copy_b128(A  + (size_t)(bm0 + srow) * K + kk + (sch0 + i) * 8,
                                &sA[nxt][srow * LDA_ + (sch0 + i) * 8]);
                async_copy_b128(Bt + (size_t)(bn0 + srow) * K + kk + (sch0 + i) * 8,
                                &sB[nxt][srow * LDA_ + (sch0 + i) * 8]);
            }
        }

        // ---- two K=32 WMMA steps per BK=64 ----
#pragma unroll
        for (int kc = 0; kc < 2; ++kc) {
            Frag fa[4], fb[2];
#pragma unroll
            for (int tm = 0; tm < 4; ++tm) {
                // 16-bit A layout: lane m = l16; e[0..7] = K 8*half+0..7,
                //                  e[8..15] = K 16+8*half+0..7
                const __bf16* p = &sA[cur][(wm + tm * 16 + l16) * LDA_ + kc * 32 + 8 * half];
                fa[tm].q[0] = *(const uint4*)p;
                fa[tm].q[1] = *(const uint4*)(p + 16);
            }
#pragma unroll
            for (int tn = 0; tn < 2; ++tn) {
                // 16-bit B layout: lane n = l16; e[0..7] = K 8*half+0..7,
                //                  e[8..15] = K 16+8*half+0..7
                const __bf16* p = &sB[cur][(wn + tn * 16 + l16) * LDA_ + kc * 32 + 8 * half];
                fb[tn].q[0] = *(const uint4*)p;
                fb[tn].q[1] = *(const uint4*)(p + 16);
            }
#pragma unroll
            for (int tm = 0; tm < 4; ++tm)
#pragma unroll
                for (int tn = 0; tn < 2; ++tn)
                    acc[tm][tn] = __builtin_amdgcn_wmma_f32_16x16x32_bf16(
                        false, fa[tm].v, false, fb[tn].v,
                        (short)0, acc[tm][tn], false, false);
        }
    }

    // ---- epilogue: C layout lane n = l16, rows r + 8*half ----
    float lsum = 0.0f;
#pragma unroll
    for (int tn = 0; tn < 2; ++tn) {
        const int gn = bn0 + wn + tn * 16 + l16;
        const float bv = bias[gn];
#pragma unroll
        for (int tm = 0; tm < 4; ++tm) {
            const int gm0 = bm0 + wm + tm * 16 + 8 * half;
#pragma unroll
            for (int r = 0; r < 8; ++r) {
                const int gm = gm0 + r;
                float x = acc[tm][tn][r] + bv;
                if (MODE == 0) {
                    outBf[(size_t)gm * N + gn] = (__bf16)gelu_exact(x);
                } else if (MODE == 1) {
                    outF[(size_t)gm * N + gn] = x;
                } else {
                    const int row = idx[gm];
                    const float* src =
                        (row < NFLAT) ? (hid + (size_t)row * HD) : (mem + (size_t)row * HD);
                    const float d = x - src[gn];
                    lsum += d * d;
                }
            }
        }
    }
    if (MODE == 2) {
#pragma unroll
        for (int off = 16; off; off >>= 1) lsum += __shfl_xor(lsum, off, 32);
        if (lane == 0) atomicAdd(accum, lsum);
    }
}

// =====================================================================
// reparameterize + KL partial sums.
// =====================================================================
__global__ __launch_bounds__(256) void reparam_kl_kernel(
    const float* __restrict__ enc, const float* __restrict__ eps,
    __bf16* __restrict__ z, float* __restrict__ klacc) {
    const int s = blockIdx.x;
    const int t = threadIdx.x;
    const float4 mu = ((const float4*)(enc + (size_t)s * (2 * COMP_)))[t];
    const float4 lv = ((const float4*)(enc + (size_t)s * (2 * COMP_) + COMP_))[t];
    const float4 ep = ((const float4*)(eps + (size_t)s * COMP_))[t];

    float z0 = mu.x + __expf(0.5f * lv.x) * ep.x;
    float z1 = mu.y + __expf(0.5f * lv.y) * ep.y;
    float z2 = mu.z + __expf(0.5f * lv.z) * ep.z;
    float z3 = mu.w + __expf(0.5f * lv.w) * ep.w;
    v4bf o = { (__bf16)z0, (__bf16)z1, (__bf16)z2, (__bf16)z3 };
    *(v4bf*)(z + (size_t)s * COMP_ + t * 4) = o;

    float kl = (1.0f + lv.x - mu.x * mu.x - __expf(lv.x))
             + (1.0f + lv.y - mu.y * mu.y - __expf(lv.y))
             + (1.0f + lv.z - mu.z * mu.z - __expf(lv.z))
             + (1.0f + lv.w - mu.w * mu.w - __expf(lv.w));
#pragma unroll
    for (int off = 16; off; off >>= 1) kl += __shfl_xor(kl, off, 32);
    if ((t & 31) == 0) atomicAdd(klacc, kl);
}

// =====================================================================
// out = recon_sum / (NS*HD) + 0.001 * ( -0.5 * kl_sum )
// =====================================================================
__global__ void finalize_kernel(const float* acc, float* out) {
    out[0] = acc[0] / ((float)NS * (float)HD) + 0.001f * (-0.5f * acc[1]);
}

// =====================================================================
extern "C" void kernel_launch(void* const* d_in, const int* in_sizes, int n_in,
                              void* d_out, int out_size, void* d_ws, size_t ws_size,
                              hipStream_t stream) {
    (void)in_sizes; (void)n_in; (void)out_size; (void)ws_size;
    const float* hid = (const float*)d_in[0];    // (4,2048,4096) -> flat 8192x4096
    const float* mem = (const float*)d_in[1];    // (32768,4096)
    const int*   idx = (const int*)d_in[2];      // (16384,)
    const float* eps = (const float*)d_in[3];    // (16384,1024)
    const float* W1  = (const float*)d_in[4];    // (4096,2048)
    const float* b1  = (const float*)d_in[5];
    const float* W2  = (const float*)d_in[6];    // (2048,2048)
    const float* b2  = (const float*)d_in[7];
    const float* Wd1 = (const float*)d_in[8];    // (1024,2048)
    const float* bd1 = (const float*)d_in[9];
    const float* Wd2 = (const float*)d_in[10];   // (2048,4096)
    const float* bd2 = (const float*)d_in[11];
    float* out = (float*)d_out;

    char* ws = (char*)d_ws;
    __bf16* samp = (__bf16*)(ws + OFF_SAMP);
    __bf16* w1t  = (__bf16*)(ws + OFF_W1T);
    __bf16* w2t  = (__bf16*)(ws + OFF_W2T);
    __bf16* wd1t = (__bf16*)(ws + OFF_WD1T);
    __bf16* wd2t = (__bf16*)(ws + OFF_WD2T);
    __bf16* hbuf = (__bf16*)(ws + OFF_H);
    float*  enc  = (float*)(ws + OFF_ENC);
    __bf16* zbuf = (__bf16*)(ws + OFF_Z);
    __bf16* hdb  = (__bf16*)(ws + OFF_HD_);
    float*  acc  = (float*)(ws + OFF_ACC);

    zero_acc_kernel<<<1, 32, 0, stream>>>(acc);

    // weight transposes to bf16 [N][K]
    transpose_bf16_kernel<<<dim3(HALF_ / 32, HD / 32),    256, 0, stream>>>(W1,  w1t,  HD,    HALF_);
    transpose_bf16_kernel<<<dim3(HALF_ / 32, HALF_ / 32), 256, 0, stream>>>(W2,  w2t,  HALF_, HALF_);
    transpose_bf16_kernel<<<dim3(HALF_ / 32, COMP_ / 32), 256, 0, stream>>>(Wd1, wd1t, COMP_, HALF_);
    transpose_bf16_kernel<<<dim3(HD / 32,    HALF_ / 32), 256, 0, stream>>>(Wd2, wd2t, HALF_, HD);

    // gather samples -> bf16
    gather_samples_kernel<<<dim3(HD / 1024, NS), 256, 0, stream>>>(hid, mem, idx, samp);

    // GEMM1: h = gelu(samples @ W1 + b1)        [NS x 2048], K=4096
    gemm_wmma_kernel<0><<<dim3(HALF_ / 128, NS / 128), 256, 0, stream>>>(
        samp, w1t, b1, NS, HALF_, HD, hbuf, nullptr, nullptr, nullptr, nullptr, nullptr);

    // GEMM2: enc = h @ W2 + b2 (fp32)           [NS x 2048], K=2048
    gemm_wmma_kernel<1><<<dim3(HALF_ / 128, NS / 128), 256, 0, stream>>>(
        hbuf, w2t, b2, NS, HALF_, HALF_, nullptr, enc, nullptr, nullptr, nullptr, nullptr);

    // reparam + KL
    reparam_kl_kernel<<<NS, 256, 0, stream>>>(enc, eps, zbuf, acc + 1);

    // GEMM3: hd = gelu(z @ Wd1 + bd1)           [NS x 2048], K=1024
    gemm_wmma_kernel<0><<<dim3(HALF_ / 128, NS / 128), 256, 0, stream>>>(
        zbuf, wd1t, bd1, NS, HALF_, COMP_, hdb, nullptr, nullptr, nullptr, nullptr, nullptr);

    // GEMM4: recon loss fused — sum((hd @ Wd2 + bd2 - samples_ref)^2)
    gemm_wmma_kernel<2><<<dim3(HD / 128, NS / 128), 256, 0, stream>>>(
        hdb, wd2t, bd2, NS, HD, HALF_, nullptr, nullptr, idx, hid, mem, acc);

    finalize_kernel<<<1, 1, 0, stream>>>(acc, out);
}
